// StochasticEnsemble_83648783056939
// MI455X (gfx1250) — matrix operand order repaired and checked by reading, not compiled, not run on hardware
//
#include <hip/hip_runtime.h>
#include <hip/hip_bf16.h>

#define N_EXPERTS 8
#define Z_DIM     128
#define HID       1024
#define OUT_DIM   3072
#define BATCH     2048
#define TMP       64            // rows/block, packed path
#define TMF       32            // rows/block, fallback path
#define BN        128           // cols/block (8 waves x 16)
#define BP        2560          // padded sorted-row capacity (2048 + 8*63 -> 64-mult)
#define NT1       (HID / 16)    // 64  n-tiles, layer 1
#define KB1       (Z_DIM / 32)  // 4   k-steps, layer 1
#define NT2       (OUT_DIM / 16)// 192 n-tiles, layer 2
#define KB2       (HID / 32)    // 32  k-steps, layer 2 (even)

typedef __attribute__((ext_vector_type(16))) __bf16 v16bf;
typedef __attribute__((ext_vector_type(8)))  float  v8f;

static __device__ __forceinline__ unsigned short f32_bf16(float f) {
  unsigned u = __float_as_uint(f);
  u += 0x7FFFu + ((u >> 16) & 1u);       // round-to-nearest-even
  return (unsigned short)(u >> 16);
}
static __device__ __forceinline__ unsigned pack2(float lo, float hi) {
  return (unsigned)f32_bf16(lo) | ((unsigned)f32_bf16(hi) << 16);
}

// gfx1250 async memory->LDS copy (no VGPR transit, tracked by ASYNCcnt)
static __device__ __forceinline__ void async_ld_b128(unsigned lds_off, const void* g) {
  asm volatile("global_load_async_to_lds_b128 %0, %1, off"
               :: "v"(lds_off), "v"((unsigned long long)(uintptr_t)g)
               : "memory");
}
static __device__ __forceinline__ void wait_async0() {
  asm volatile("s_wait_asynccnt 0x0" ::: "memory");
}

// ---------------- routing ----------------

__global__ void k_init(int* cnt) {
  if (threadIdx.x < N_EXPERTS) cnt[threadIdx.x] = 0;
}

__global__ void k_route(const float* __restrict__ z, int* cnt,
                        int* eIdx, int* ePos) {
  int samp = (blockIdx.x * blockDim.x + threadIdx.x) >> 5;   // one wave32 / sample
  int lane = threadIdx.x & 31;
  if (samp >= BATCH) return;
  const float* zr = z + (size_t)samp * Z_DIM;
  float s = 0.f;
  #pragma unroll
  for (int k = lane; k < Z_DIM; k += 32)
    s += floorf(fabsf(zr[k]) * 1000.0f);
  #pragma unroll
  for (int off = 16; off; off >>= 1)
    s += __shfl_xor(s, off, 32);
  if (lane == 0) {
    int e = ((int)s) & (N_EXPERTS - 1);
    int p = atomicAdd(&cnt[e], 1);
    eIdx[samp] = e;
    ePos[samp] = p;
  }
}

__global__ void k_offsets(const int* __restrict__ cnt, int* poff, int* rowmap) {
  if (threadIdx.x == 0) {
    int acc = 0;
    #pragma unroll
    for (int e = 0; e < N_EXPERTS; ++e) {
      poff[e] = acc;
      acc += (cnt[e] + (TMP - 1)) & ~(TMP - 1);   // pad each expert to 64 rows
    }
    poff[N_EXPERTS] = acc;
  }
  for (int r = threadIdx.x; r < BP; r += blockDim.x) rowmap[r] = -1;
}

__global__ void k_scatter(const int* __restrict__ eIdx, const int* __restrict__ ePos,
                          const int* __restrict__ poff, int* rowmap) {
  int i = blockIdx.x * blockDim.x + threadIdx.x;
  if (i >= BATCH) return;
  rowmap[poff[eIdx[i]] + ePos[i]] = i;
}

// -------- weight pre-pack: f32 [e][K][N] -> bf16 WMMA-B-fragment layout --------
// Wp[((e*NT + nt)*KB + kb)*32 + lane][16]:
//   lanes 0-15 hold K = kb*32+{0..7,16..23} at col nt*16+(lane&15); lanes 16-31 hold K+8.

__global__ void k_pack(const float* __restrict__ W, unsigned short* __restrict__ Wp,
                       int K, int N, int total32) {
  int tid = blockIdx.x * blockDim.x + threadIdx.x;
  if (tid >= total32) return;
  int NT = N >> 4, KB = K >> 5;
  int lane = tid & 31;
  int g    = tid >> 5;
  int kb   = g % KB;
  int nt   = (g / KB) % NT;
  int e    = g / (KB * NT);
  int n  = nt * 16 + (lane & 15);
  int k0 = kb * 32 + ((lane & 16) ? 8 : 0);
  const float* src = W + ((size_t)e * K + k0) * N + n;
  unsigned short tmp[16];
  #pragma unroll
  for (int j = 0; j < 16; ++j)
    tmp[j] = f32_bf16(src[(size_t)(j < 8 ? j : j + 8) * N]);
  uint4* dst = (uint4*)&Wp[(size_t)tid * 16];
  dst[0] = *(uint4*)&tmp[0];
  dst[1] = *(uint4*)&tmp[8];
}

// ============ packed-weight GEMMs: TM=64, B fragments straight from global ============

__global__ void __launch_bounds__(256)
kp_gemm1(const float* __restrict__ z, const unsigned short* __restrict__ Wp1,
         const float* __restrict__ b1, const int* __restrict__ poff,
         const int* __restrict__ rowmap, unsigned short* __restrict__ Hs) {
  __shared__ __attribute__((aligned(16))) unsigned short sA[TMP][32];

  const int rowBase = blockIdx.y * TMP;
  if (rowBase >= poff[N_EXPERTS]) return;
  int e = 0;
  while (e < N_EXPERTS - 1 && rowBase >= poff[e + 1]) ++e;

  const int t     = threadIdx.x;
  const int lane  = t & 31;
  const int ntile = blockIdx.x * 8 + (t >> 5);

  // A staging: thread owns (row r, 8-wide k chunk kc), constant across k-steps
  const int r  = t >> 2;
  const int kc = (t & 3) * 8;
  int s = rowmap[rowBase + r];
  const float* zr = z + (size_t)(s < 0 ? 0 : s) * Z_DIM + kc;   // clamp: pad rows discarded later

  const unsigned short* bp =
      Wp1 + ((((size_t)e * NT1 + ntile) * KB1) * 32 + lane) * 16;

  v8f acc0 = {}, acc1 = {}, acc2 = {}, acc3 = {};
  const int klo  = (lane < 16) ? 0 : 8;
  const int arow = lane & 15;

  #pragma unroll
  for (int kb = 0; kb < KB1; ++kb) {
    __syncthreads();
    {  // stage 8 bf16 of A
      const float* p = zr + kb * 32;
      float4 f0 = *(const float4*)(p);
      float4 f1 = *(const float4*)(p + 4);
      uint4 pk;
      pk.x = pack2(f0.x, f0.y); pk.y = pack2(f0.z, f0.w);
      pk.z = pack2(f1.x, f1.y); pk.w = pack2(f1.z, f1.w);
      *(uint4*)&sA[r][kc] = pk;
    }
    union { v16bf v; uint4 q[2]; } b;
    {
      const uint4* q = (const uint4*)(bp + (size_t)kb * 512);
      b.q[0] = q[0]; b.q[1] = q[1];
    }
    __syncthreads();
    union { v16bf v; uint4 q[2]; } a0, a1, a2, a3;
    a0.q[0] = *(const uint4*)&sA[arow][klo];      a0.q[1] = *(const uint4*)&sA[arow][klo + 16];
    a1.q[0] = *(const uint4*)&sA[16 + arow][klo]; a1.q[1] = *(const uint4*)&sA[16 + arow][klo + 16];
    a2.q[0] = *(const uint4*)&sA[32 + arow][klo]; a2.q[1] = *(const uint4*)&sA[32 + arow][klo + 16];
    a3.q[0] = *(const uint4*)&sA[48 + arow][klo]; a3.q[1] = *(const uint4*)&sA[48 + arow][klo + 16];
    acc0 = __builtin_amdgcn_wmma_f32_16x16x32_bf16(false, a0.v, false, b.v, (short)0, acc0, false, false);
    acc1 = __builtin_amdgcn_wmma_f32_16x16x32_bf16(false, a1.v, false, b.v, (short)0, acc1, false, false);
    acc2 = __builtin_amdgcn_wmma_f32_16x16x32_bf16(false, a2.v, false, b.v, (short)0, acc2, false, false);
    acc3 = __builtin_amdgcn_wmma_f32_16x16x32_bf16(false, a3.v, false, b.v, (short)0, acc3, false, false);
  }

  const int nc   = lane & 15;
  const int nCol = ntile * 16 + nc;
  const float bv = b1[(size_t)e * HID + nCol];
  #pragma unroll
  for (int i = 0; i < 8; ++i) {
    int m = (lane < 16) ? i : (i + 8);
    float v0 = acc0[i] + bv; v0 = v0 > 0.f ? v0 : 0.f;
    float v1 = acc1[i] + bv; v1 = v1 > 0.f ? v1 : 0.f;
    float v2 = acc2[i] + bv; v2 = v2 > 0.f ? v2 : 0.f;
    float v3 = acc3[i] + bv; v3 = v3 > 0.f ? v3 : 0.f;
    Hs[(size_t)(rowBase + m)      * HID + nCol] = f32_bf16(v0);
    Hs[(size_t)(rowBase + 16 + m) * HID + nCol] = f32_bf16(v1);
    Hs[(size_t)(rowBase + 32 + m) * HID + nCol] = f32_bf16(v2);
    Hs[(size_t)(rowBase + 48 + m) * HID + nCol] = f32_bf16(v3);
  }
}

__global__ void __launch_bounds__(256)
kp_gemm2(const unsigned short* __restrict__ Hs, const unsigned short* __restrict__ Wp2,
         const float* __restrict__ b2, const int* __restrict__ poff,
         const int* __restrict__ rowmap, float* __restrict__ out) {
  __shared__ __attribute__((aligned(16))) unsigned short sA[2][TMP][32]; // double buffer

  const int rowBase = blockIdx.y * TMP;
  if (rowBase >= poff[N_EXPERTS]) return;
  int e = 0;
  while (e < N_EXPERTS - 1 && rowBase >= poff[e + 1]) ++e;

  const int t     = threadIdx.x;
  const int lane  = t & 31;
  const int ntile = blockIdx.x * 8 + (t >> 5);

  const int r  = t >> 2;
  const int kc = (t & 3) * 8;
  const unsigned short* hr = Hs + (size_t)(rowBase + r) * HID + kc;
  const unsigned lds0 = (unsigned)(uintptr_t)&sA[0][r][kc];
  const unsigned lds1 = (unsigned)(uintptr_t)&sA[1][r][kc];

  const unsigned short* bp =
      Wp2 + ((((size_t)e * NT2 + ntile) * KB2) * 32 + lane) * 16;

  v8f acc0 = {}, acc1 = {}, acc2 = {}, acc3 = {};
  const int klo  = (lane < 16) ? 0 : 8;
  const int arow = lane & 15;

#define MMA_STEP(BUF, KB_IDX)                                                            \
  {                                                                                      \
    union { v16bf v; uint4 q[2]; } b;                                                    \
    const uint4* qq = (const uint4*)(bp + (size_t)(KB_IDX) * 512);                       \
    b.q[0] = qq[0]; b.q[1] = qq[1];                                                      \
    union { v16bf v; uint4 q[2]; } a0, a1, a2, a3;                                       \
    a0.q[0] = *(const uint4*)&sA[BUF][arow][klo];                                        \
    a0.q[1] = *(const uint4*)&sA[BUF][arow][klo + 16];                                   \
    a1.q[0] = *(const uint4*)&sA[BUF][16 + arow][klo];                                   \
    a1.q[1] = *(const uint4*)&sA[BUF][16 + arow][klo + 16];                              \
    a2.q[0] = *(const uint4*)&sA[BUF][32 + arow][klo];                                   \
    a2.q[1] = *(const uint4*)&sA[BUF][32 + arow][klo + 16];                              \
    a3.q[0] = *(const uint4*)&sA[BUF][48 + arow][klo];                                   \
    a3.q[1] = *(const uint4*)&sA[BUF][48 + arow][klo + 16];                              \
    acc0 = __builtin_amdgcn_wmma_f32_16x16x32_bf16(false, a0.v, false, b.v, (short)0, acc0, false, false); \
    acc1 = __builtin_amdgcn_wmma_f32_16x16x32_bf16(false, a1.v, false, b.v, (short)0, acc1, false, false); \
    acc2 = __builtin_amdgcn_wmma_f32_16x16x32_bf16(false, a2.v, false, b.v, (short)0, acc2, false, false); \
    acc3 = __builtin_amdgcn_wmma_f32_16x16x32_bf16(false, a3.v, false, b.v, (short)0, acc3, false, false); \
  }

  // prologue: async-stage k-step 0 into buffer 0
  async_ld_b128(lds0, hr);
  wait_async0();
  __syncthreads();

  // steady state, unrolled x2 with explicit buffers (KB2 even)
  for (int kb = 0; kb < KB2 - 2; kb += 2) {
    // compute kb from buf0, async-stage kb+1 into buf1
    async_ld_b128(lds1, hr + (size_t)(kb + 1) * 32);
    __builtin_prefetch(bp + (size_t)(kb + 1) * 512, 0, 1);
    MMA_STEP(0, kb);
    wait_async0();
    __syncthreads();
    // compute kb+1 from buf1, async-stage kb+2 into buf0
    async_ld_b128(lds0, hr + (size_t)(kb + 2) * 32);
    __builtin_prefetch(bp + (size_t)(kb + 2) * 512, 0, 1);
    MMA_STEP(1, kb + 1);
    wait_async0();
    __syncthreads();
  }
  // epilogue: last two k-steps
  async_ld_b128(lds1, hr + (size_t)(KB2 - 1) * 32);
  MMA_STEP(0, KB2 - 2);
  wait_async0();
  __syncthreads();
  MMA_STEP(1, KB2 - 1);
#undef MMA_STEP

  const int nc   = lane & 15;
  const int nCol = ntile * 16 + nc;
  const float bv = b2[(size_t)e * OUT_DIM + nCol];
  #pragma unroll
  for (int i = 0; i < 8; ++i) {
    int m = (lane < 16) ? i : (i + 8);
    int s0 = rowmap[rowBase + m];
    int s1 = rowmap[rowBase + 16 + m];
    int s2 = rowmap[rowBase + 32 + m];
    int s3 = rowmap[rowBase + 48 + m];
    if (s0 >= 0) out[(size_t)s0 * OUT_DIM + nCol] = acc0[i] + bv;
    if (s1 >= 0) out[(size_t)s1 * OUT_DIM + nCol] = acc1[i] + bv;
    if (s2 >= 0) out[(size_t)s2 * OUT_DIM + nCol] = acc2[i] + bv;
    if (s3 >= 0) out[(size_t)s3 * OUT_DIM + nCol] = acc3[i] + bv;
  }
}

// ============ fallback GEMMs (small ws): TM=32, LDS-staged B ============

__global__ void __launch_bounds__(256)
kf_gemm1(const float* __restrict__ z, const float* __restrict__ W1,
         const float* __restrict__ b1, const int* __restrict__ poff,
         const int* __restrict__ rowmap, unsigned short* __restrict__ Hs) {
  __shared__ __attribute__((aligned(16))) unsigned short sA[TMF][32];
  __shared__ __attribute__((aligned(16))) unsigned short sB[BN][32];

  const int rowBase = blockIdx.y * TMF;
  if (rowBase >= poff[N_EXPERTS]) return;
  int e = 0;
  while (e < N_EXPERTS - 1 && rowBase >= poff[e + 1]) ++e;

  const int nBlock = blockIdx.x * BN;
  const int t = threadIdx.x, lane = t & 31, waveN = (t >> 5) * 16;
  v8f acc0 = {}, acc1 = {};

  for (int kb = 0; kb < Z_DIM; kb += 32) {
    __syncthreads();
    #pragma unroll
    for (int i = 0; i < (TMF * 32) / 256; ++i) {
      int linear = i * 256 + t;
      int rr = linear >> 5, k = linear & 31;
      int s = rowmap[rowBase + rr];
      sA[rr][k] = f32_bf16(z[(size_t)(s < 0 ? 0 : s) * Z_DIM + kb + k]);
    }
    const float* Wt = W1 + ((size_t)e * Z_DIM + kb) * HID + nBlock;
    #pragma unroll
    for (int i = 0; i < (32 * BN) / 256; ++i) {
      int linear = i * 256 + t;
      int n = linear & (BN - 1), k = linear >> 7;
      sB[n][k] = f32_bf16(Wt[(size_t)k * HID + n]);
    }
    __syncthreads();
    union { v16bf v; uint4 q[2]; } a0, a1, b;
    const int klo = (lane < 16) ? 0 : 8, arow = lane & 15;
    a0.q[0] = *(const uint4*)&sA[arow][klo];      a0.q[1] = *(const uint4*)&sA[arow][klo + 16];
    a1.q[0] = *(const uint4*)&sA[16 + arow][klo]; a1.q[1] = *(const uint4*)&sA[16 + arow][klo + 16];
    const int bcol = waveN + (lane & 15), bk = (lane < 16) ? 0 : 8;
    b.q[0] = *(const uint4*)&sB[bcol][bk]; b.q[1] = *(const uint4*)&sB[bcol][bk + 16];
    acc0 = __builtin_amdgcn_wmma_f32_16x16x32_bf16(false, a0.v, false, b.v, (short)0, acc0, false, false);
    acc1 = __builtin_amdgcn_wmma_f32_16x16x32_bf16(false, a1.v, false, b.v, (short)0, acc1, false, false);
  }
  const int nc = lane & 15;
  const float bv = b1[(size_t)e * HID + nBlock + waveN + nc];
  #pragma unroll
  for (int i = 0; i < 8; ++i) {
    int m = (lane < 16) ? i : (i + 8);
    float v0 = acc0[i] + bv; v0 = v0 > 0.f ? v0 : 0.f;
    float v1 = acc1[i] + bv; v1 = v1 > 0.f ? v1 : 0.f;
    Hs[(size_t)(rowBase + m)      * HID + nBlock + waveN + nc] = f32_bf16(v0);
    Hs[(size_t)(rowBase + 16 + m) * HID + nBlock + waveN + nc] = f32_bf16(v1);
  }
}

__global__ void __launch_bounds__(256)
kf_gemm2(const unsigned short* __restrict__ Hs, const float* __restrict__ W2,
         const float* __restrict__ b2, const int* __restrict__ poff,
         const int* __restrict__ rowmap, float* __restrict__ out) {
  __shared__ __attribute__((aligned(16))) unsigned short sA[TMF][32];
  __shared__ __attribute__((aligned(16))) unsigned short sB[BN][32];

  const int rowBase = blockIdx.y * TMF;
  if (rowBase >= poff[N_EXPERTS]) return;
  int e = 0;
  while (e < N_EXPERTS - 1 && rowBase >= poff[e + 1]) ++e;

  const int nBlock = blockIdx.x * BN;
  const int t = threadIdx.x, lane = t & 31, waveN = (t >> 5) * 16;
  v8f acc0 = {}, acc1 = {};

  for (int kb = 0; kb < HID; kb += 32) {
    __syncthreads();
    #pragma unroll
    for (int i = 0; i < (TMF * 32) / 256; ++i) {
      int linear = i * 256 + t;
      int rr = linear >> 5, k = linear & 31;
      sA[rr][k] = Hs[(size_t)(rowBase + rr) * HID + kb + k];
    }
    const float* Wt = W2 + ((size_t)e * HID + kb) * OUT_DIM + nBlock;
    #pragma unroll
    for (int i = 0; i < (32 * BN) / 256; ++i) {
      int linear = i * 256 + t;
      int n = linear & (BN - 1), k = linear >> 7;
      sB[n][k] = f32_bf16(Wt[(size_t)k * OUT_DIM + n]);
    }
    __syncthreads();
    union { v16bf v; uint4 q[2]; } a0, a1, b;
    const int klo = (lane < 16) ? 0 : 8, arow = lane & 15;
    a0.q[0] = *(const uint4*)&sA[arow][klo];      a0.q[1] = *(const uint4*)&sA[arow][klo + 16];
    a1.q[0] = *(const uint4*)&sA[16 + arow][klo]; a1.q[1] = *(const uint4*)&sA[16 + arow][klo + 16];
    const int bcol = waveN + (lane & 15), bk = (lane < 16) ? 0 : 8;
    b.q[0] = *(const uint4*)&sB[bcol][bk]; b.q[1] = *(const uint4*)&sB[bcol][bk + 16];
    acc0 = __builtin_amdgcn_wmma_f32_16x16x32_bf16(false, a0.v, false, b.v, (short)0, acc0, false, false);
    acc1 = __builtin_amdgcn_wmma_f32_16x16x32_bf16(false, a1.v, false, b.v, (short)0, acc1, false, false);
  }
  const int nc = lane & 15, nCol = nBlock + waveN + nc;
  const float bv = b2[(size_t)e * OUT_DIM + nCol];
  #pragma unroll
  for (int i = 0; i < 8; ++i) {
    int m = (lane < 16) ? i : (i + 8);
    int s0 = rowmap[rowBase + m];
    int s1 = rowmap[rowBase + 16 + m];
    if (s0 >= 0) out[(size_t)s0 * OUT_DIM + nCol] = acc0[i] + bv;
    if (s1 >= 0) out[(size_t)s1 * OUT_DIM + nCol] = acc1[i] + bv;
  }
}

// ---------------- launcher ----------------

extern "C" void kernel_launch(void* const* d_in, const int* in_sizes, int n_in,
                              void* d_out, int out_size, void* d_ws, size_t ws_size,
                              hipStream_t stream) {
  const float* z  = (const float*)d_in[0];
  const float* W1 = (const float*)d_in[1];
  const float* b1 = (const float*)d_in[2];
  const float* W2 = (const float*)d_in[3];
  const float* b2 = (const float*)d_in[4];
  float* out = (float*)d_out;

  char* ws = (char*)d_ws;
  size_t off = 0;
  int* cnt    = (int*)(ws + off); off += 64;
  int* poff   = (int*)(ws + off); off += 64;
  int* eIdx   = (int*)(ws + off); off += 4 * BATCH;
  int* ePos   = (int*)(ws + off); off += 4 * BATCH;
  int* rowmap = (int*)(ws + off); off += 4 * BP;
  off = (off + 255) & ~(size_t)255;
  unsigned short* Hs = (unsigned short*)(ws + off); off += (size_t)BP * HID * 2;
  off = (off + 255) & ~(size_t)255;
  unsigned short* Wp1 = (unsigned short*)(ws + off); off += (size_t)N_EXPERTS * Z_DIM * HID * 2;
  off = (off + 255) & ~(size_t)255;
  unsigned short* Wp2 = (unsigned short*)(ws + off); off += (size_t)N_EXPERTS * HID * OUT_DIM * 2;
  const size_t NEED_PACKED = off;
  const bool packed = (ws_size >= NEED_PACKED);

  k_init<<<1, 256, 0, stream>>>(cnt);
  k_route<<<(BATCH * 32) / 256, 256, 0, stream>>>(z, cnt, eIdx, ePos);
  k_offsets<<<1, 256, 0, stream>>>(cnt, poff, rowmap);
  k_scatter<<<BATCH / 256, 256, 0, stream>>>(eIdx, ePos, poff, rowmap);

  if (packed) {
    int tot1 = N_EXPERTS * NT1 * KB1 * 32;          // 65536
    int tot2 = N_EXPERTS * NT2 * KB2 * 32;          // 1572864
    k_pack<<<(tot1 + 255) / 256, 256, 0, stream>>>(W1, Wp1, Z_DIM, HID, tot1);
    k_pack<<<(tot2 + 255) / 256, 256, 0, stream>>>(W2, Wp2, HID, OUT_DIM, tot2);
    dim3 g1(HID / BN, BP / TMP);     // 8 x 40
    kp_gemm1<<<g1, 256, 0, stream>>>(z, Wp1, b1, poff, rowmap, Hs);
    dim3 g2(OUT_DIM / BN, BP / TMP); // 24 x 40
    kp_gemm2<<<g2, 256, 0, stream>>>(Hs, Wp2, b2, poff, rowmap, out);
  } else {
    dim3 g1(HID / BN, BP / TMF);     // 8 x 80
    kf_gemm1<<<g1, 256, 0, stream>>>(z, W1, b1, poff, rowmap, Hs);
    dim3 g2(OUT_DIM / BN, BP / TMF); // 24 x 80
    kf_gemm2<<<g2, 256, 0, stream>>>(Hs, W2, b2, poff, rowmap, out);
  }
}
